// RelationNetwork_68805376082627
// MI455X (gfx1250) — matrix operand order repaired and checked by reading, not compiled, not run on hardware
//
#include <hip/hip_runtime.h>
#include <hip/hip_bf16.h>

typedef __attribute__((ext_vector_type(16))) __bf16 v16bf;
typedef __attribute__((ext_vector_type(8)))  __bf16 v8bf;
typedef __attribute__((ext_vector_type(8)))  float  v8f;

#define NOBJ 512
#define DD   128
#define NWAVE 8                   // waves per block (256 threads)
#define NBLK  256                 // blocks
#define TOTWAVE (NBLK * NWAVE)    // 2048 waves
#define NTILES (NOBJ * NOBJ / 16) // 16384 tiles of 16 pairs
#define ITERS (NTILES / TOTWAVE)  // 8

__device__ __forceinline__ float relu(float v) { return v > 0.0f ? v : 0.0f; }

// -------------------------------------------------------------------------
// Kernel 1: A' = x@Wa + (q@Wc + b1),  B = x@Wb ; zero embedding accumulator
// -------------------------------------------------------------------------
__global__ __launch_bounds__(128) void rn_precompute(
    const float* __restrict__ x, const float* __restrict__ q,
    const float* __restrict__ g_w1, const float* __restrict__ g_b1,
    float* __restrict__ Ap, float* __restrict__ Bp, float* __restrict__ emb) {
  int i = blockIdx.x;      // object row 0..511
  int t = threadIdx.x;     // output feature 0..127
  if (i == 0) emb[t] = 0.0f;
  float sa = 0.f, sb = 0.f, sc = 0.f;
  #pragma unroll 8
  for (int k = 0; k < DD; ++k) {
    float xv = x[i * DD + k];
    sa += xv   * g_w1[(k)        * DD + t];   // Wa rows [0,128)
    sb += xv   * g_w1[(DD + k)   * DD + t];   // Wb rows [128,256)
    sc += q[k] * g_w1[(2*DD + k) * DD + t];   // Wc rows [256,384)
  }
  Ap[i * DD + t] = sa + sc + g_b1[t];
  Bp[i * DD + t] = sb;
}

// -------------------------------------------------------------------------
// Kernel 2: main pair loop, layers 2+3 via v_wmma_f32_16x16x32_bf16
// -------------------------------------------------------------------------
__global__ __launch_bounds__(256)
__attribute__((amdgpu_waves_per_eu(4)))   // <=256 VGPRs: 4 waves/SIMD, no vgpr-msb
void rn_main(
    const float* __restrict__ Ap, const float* __restrict__ Bp,
    const float* __restrict__ g_w2, const float* __restrict__ g_b2,
    const float* __restrict__ g_w3, const float* __restrict__ g_b3,
    float* __restrict__ emb) {
  extern __shared__ unsigned char smem[];
  __bf16* w2f   = (__bf16*)smem;                          // 8*4*32*16 bf16 = 32 KB
  __bf16* w3f   = (__bf16*)(smem + 32768);                // 32 KB
  __bf16* h2s   = (__bf16*)(smem + 65536);                // NWAVE * 16*128 bf16 = 32 KB
  float*  wgEmb = (float*)(smem + 65536 + NWAVE * 16 * DD * 2);

  const int tid  = threadIdx.x;
  const int lane = tid & 31;
  const int wv   = tid >> 5;

  // --- pre-swizzle W2/W3 into bf16 B-fragment layout in LDS ---
  // B-frag (32x16 KxN, 16-bit): lane L -> N = L&15, Kbase = (L&16)?8:0, elem e -> K = Kbase+(e&7)+((e&8)?16:0)
  for (int idx = tid; idx < 8 * 4 * 32 * 16; idx += 256) {
    int e  = idx & 15;
    int ln = (idx >> 4) & 31;
    int kc = (idx >> 9) & 3;
    int nt = idx >> 11;
    int k  = kc * 32 + ((ln & 16) ? 8 : 0) + (e & 7) + ((e & 8) ? 16 : 0);
    int n  = nt * 16 + (ln & 15);
    w2f[idx] = (__bf16)g_w2[k * DD + n];
    w3f[idx] = (__bf16)g_w3[k * DD + n];
  }
  if (tid < DD) wgEmb[tid] = 0.0f;
  __syncthreads();

  // --- per-lane constants ---
  const int colN = lane & 15;                 // C/D column == A-frag row index
  const int kb_a = (lane & 16) ? 8 : 0;       // A-frag K base (16-bit layout)
  const int rowBase = (lane & 16) ? 8 : 0;    // C/D row base
  const int wave = blockIdx.x * NWAVE + wv;   // 0..2047
  const int j0   = (wave & 31) * 16;          // fixed j-tile for this wave

  float b2v[8], b3v[8];
  #pragma unroll
  for (int nt = 0; nt < 8; ++nt) {
    b2v[nt] = g_b2[nt * 16 + colN];
    b3v[nt] = g_b3[nt * 16 + colN];
  }

  float colAcc[8];
  #pragma unroll
  for (int nt = 0; nt < 8; ++nt) colAcc[nt] = 0.0f;

  const float* Bj = Bp + (j0 + colN) * DD;
  __bf16* h2 = &h2s[wv * 16 * DD];
  const v8f vzero = {};   // inline-0 C operand; free to rematerialize, never spilled

  #pragma unroll 1
  for (int it = 0; it < ITERS; ++it) {
    const int tile = wave + it * TOTWAVE;
    const int i = tile >> 5;                  // tile&31 == wave&31 (same j-tile)
    const float* Ai = Ap + i * DD;

    // ---- build h1 = relu(A'[i] + B[j]) directly as bf16 A-fragments ----
    // A-frag elems 0..7  <-> K = kc*32 + kb_a + e        (contiguous)
    //        elems 8..15 <-> K = kc*32 + kb_a + 16 + e   (contiguous)
    v16bf a1[4];
    #pragma unroll
    for (int kc = 0; kc < 4; ++kc) {
      int K0 = kc * 32 + kb_a;
      float alo[8], ahi[8], blo[8], bhi[8];
      *(float4*)&alo[0] = *(const float4*)&Ai[K0];
      *(float4*)&alo[4] = *(const float4*)&Ai[K0 + 4];
      *(float4*)&ahi[0] = *(const float4*)&Ai[K0 + 16];
      *(float4*)&ahi[4] = *(const float4*)&Ai[K0 + 20];
      *(float4*)&blo[0] = *(const float4*)&Bj[K0];
      *(float4*)&blo[4] = *(const float4*)&Bj[K0 + 4];
      *(float4*)&bhi[0] = *(const float4*)&Bj[K0 + 16];
      *(float4*)&bhi[4] = *(const float4*)&Bj[K0 + 20];
      #pragma unroll
      for (int e = 0; e < 8; ++e) {
        a1[kc][e]     = (__bf16)relu(alo[e] + blo[e]);
        a1[kc][8 + e] = (__bf16)relu(ahi[e] + bhi[e]);
      }
    }

    // ---- layer 2: h2 = relu(h1 @ W2 + b2), staged to LDS as bf16 ----
    #pragma unroll
    for (int nt = 0; nt < 8; ++nt) {
      // batch the 4 weight-fragment loads ahead of the WMMA chain
      v16bf w0 = *(const v16bf*)&w2f[((nt * 4 + 0) * 32 + lane) * 16];
      v16bf w1 = *(const v16bf*)&w2f[((nt * 4 + 1) * 32 + lane) * 16];
      v16bf w2 = *(const v16bf*)&w2f[((nt * 4 + 2) * 32 + lane) * 16];
      v16bf w3 = *(const v16bf*)&w2f[((nt * 4 + 3) * 32 + lane) * 16];
      // C starts as inline 0; bias folded into the epilogue
      v8f acc = __builtin_amdgcn_wmma_f32_16x16x32_bf16(false, a1[0], false, w0, (short)0, vzero, false, false);
      acc = __builtin_amdgcn_wmma_f32_16x16x32_bf16(false, a1[1], false, w1, (short)0, acc, false, false);
      acc = __builtin_amdgcn_wmma_f32_16x16x32_bf16(false, a1[2], false, w2, (short)0, acc, false, false);
      acc = __builtin_amdgcn_wmma_f32_16x16x32_bf16(false, a1[3], false, w3, (short)0, acc, false, false);
      int cc = nt * 16 + colN;
      #pragma unroll
      for (int r = 0; r < 8; ++r)
        h2[(rowBase + r) * DD + cc] = (__bf16)relu(acc[r] + b2v[nt]);
    }
    __builtin_amdgcn_wave_barrier();  // per-wave staging: DS ops are in-order in-wave

    // ---- rebuild A-fragments for layer 3 from staged h2 (2x16B LDS loads) ----
    v16bf a2[4];
    #pragma unroll
    for (int kc = 0; kc < 4; ++kc) {
      int K0 = kc * 32 + kb_a;
      v8bf lo = *(const v8bf*)&h2[colN * DD + K0];
      v8bf hi = *(const v8bf*)&h2[colN * DD + K0 + 16];
      #pragma unroll
      for (int e = 0; e < 8; ++e) {
        a2[kc][e]     = lo[e];
        a2[kc][8 + e] = hi[e];
      }
    }

    // ---- layer 3: relations = relu(h2 @ W3 + b3); accumulate column sums ----
    #pragma unroll
    for (int nt = 0; nt < 8; ++nt) {
      v16bf w0 = *(const v16bf*)&w3f[((nt * 4 + 0) * 32 + lane) * 16];
      v16bf w1 = *(const v16bf*)&w3f[((nt * 4 + 1) * 32 + lane) * 16];
      v16bf w2 = *(const v16bf*)&w3f[((nt * 4 + 2) * 32 + lane) * 16];
      v16bf w3v = *(const v16bf*)&w3f[((nt * 4 + 3) * 32 + lane) * 16];
      v8f acc = __builtin_amdgcn_wmma_f32_16x16x32_bf16(false, a2[0], false, w0, (short)0, vzero, false, false);
      acc = __builtin_amdgcn_wmma_f32_16x16x32_bf16(false, a2[1], false, w1, (short)0, acc, false, false);
      acc = __builtin_amdgcn_wmma_f32_16x16x32_bf16(false, a2[2], false, w2, (short)0, acc, false, false);
      acc = __builtin_amdgcn_wmma_f32_16x16x32_bf16(false, a2[3], false, w3v, (short)0, acc, false, false);
      float s = 0.0f;
      #pragma unroll
      for (int r = 0; r < 8; ++r) s += relu(acc[r] + b3v[nt]);
      colAcc[nt] += s;
    }
    __builtin_amdgcn_wave_barrier();  // WAR protection on h2 staging (in-wave order)
  }

  // --- reduce: lane pair (L, L+16) both add into column nt*16 + (L&15) ---
  #pragma unroll
  for (int nt = 0; nt < 8; ++nt)
    atomicAdd(&wgEmb[nt * 16 + colN], colAcc[nt]);
  __syncthreads();
  if (tid < DD) atomicAdd(&emb[tid], wgEmb[tid]);
}

// -------------------------------------------------------------------------
// Kernel 3: f-MLP on the 128-dim embedding
// -------------------------------------------------------------------------
__global__ __launch_bounds__(128) void rn_fmlp(
    const float* __restrict__ emb,
    const float* __restrict__ f_w1, const float* __restrict__ f_b1,
    const float* __restrict__ f_w2, const float* __restrict__ f_b2,
    float* __restrict__ out) {
  __shared__ float e[DD];
  __shared__ float h[DD];
  int t = threadIdx.x;
  e[t] = emb[t];
  __syncthreads();
  float s = f_b1[t];
  #pragma unroll 8
  for (int k = 0; k < DD; ++k) s += e[k] * f_w1[k * DD + t];
  h[t] = relu(s);
  __syncthreads();
  float o = f_b2[t];
  #pragma unroll 8
  for (int k = 0; k < DD; ++k) o += h[k] * f_w2[k * DD + t];
  out[t] = o;
}

// -------------------------------------------------------------------------
extern "C" void kernel_launch(void* const* d_in, const int* in_sizes, int n_in,
                              void* d_out, int out_size, void* d_ws, size_t ws_size,
                              hipStream_t stream) {
  const float* x    = (const float*)d_in[0];
  const float* q    = (const float*)d_in[1];
  const float* g_w1 = (const float*)d_in[2];
  const float* g_b1 = (const float*)d_in[3];
  const float* g_w2 = (const float*)d_in[4];
  const float* g_b2 = (const float*)d_in[5];
  const float* g_w3 = (const float*)d_in[6];
  const float* g_b3 = (const float*)d_in[7];
  const float* f_w1 = (const float*)d_in[8];
  const float* f_b1 = (const float*)d_in[9];
  const float* f_w2 = (const float*)d_in[10];
  const float* f_b2 = (const float*)d_in[11];

  float* ws  = (float*)d_ws;
  float* Ap  = ws;                    // 512*128 f32
  float* Bp  = ws + NOBJ * DD;        // 512*128 f32
  float* emb = ws + 2 * NOBJ * DD;    // 128 f32

  rn_precompute<<<NOBJ, DD, 0, stream>>>(x, q, g_w1, g_b1, Ap, Bp, emb);

  size_t lds = 32768 /*w2f*/ + 32768 /*w3f*/ + NWAVE * 16 * DD * 2 /*h2 bf16*/ + DD * 4 /*wgEmb*/;
  rn_main<<<NBLK, 256, lds, stream>>>(Ap, Bp, g_w2, g_b2, g_w3, g_b3, emb);

  rn_fmlp<<<1, DD, 0, stream>>>(emb, f_w1, f_b1, f_w2, f_b2, (float*)d_out);
}
